// CNN_LeNetSym_56959856279940
// MI455X (gfx1250) — matrix operand order repaired and checked by reading, not compiled, not run on hardware
//
#include <hip/hip_runtime.h>
#include <math.h>

typedef __attribute__((ext_vector_type(16))) _Float16 v16h;
typedef __attribute__((ext_vector_type(8)))  _Float16 v8h;
typedef __attribute__((ext_vector_type(8)))  float    v8f;

#define NK   512           // N_CLUSTERS
#define BATCH 4096

// ---------------------------------------------------------------------------
// 1) Discretize: nearest scalar centroid (brute force over 512, centroids in LDS)
// ---------------------------------------------------------------------------
__global__ void discretize_kernel(const float* __restrict__ x,
                                  const float* __restrict__ cent,
                                  unsigned short* __restrict__ sym, int n)
{
    __shared__ float c[NK];
    for (int i = threadIdx.x; i < NK; i += blockDim.x) c[i] = cent[i];
    __syncthreads();
    int idx = blockIdx.x * blockDim.x + threadIdx.x;
    if (idx >= n) return;
    float v = x[idx];
    float best = fabsf(v - c[0]);
    int bi = 0;
    #pragma unroll 4
    for (int k = 1; k < NK; ++k) {
        float d = fabsf(v - c[k]);
        if (d < best) { best = d; bi = k; }
    }
    sym[idx] = (unsigned short)bi;
}

// ---------------------------------------------------------------------------
// 2) Conv1 (k=5,s=2, C=1 -> 6ch) symbolic: LUT product, sort-25, add-LUT fold
//    One thread = one (b, oy, ox) window, loops 6 output channels.
//    Per-thread sort column in LDS: arr[i*64 + tid] -> bank-conflict free.
// ---------------------------------------------------------------------------
#define C1_BLOCK 64
__global__ void conv1_kernel(const unsigned short* __restrict__ sym,   // [B,32,32]
                             const int* __restrict__ c1w,              // [25,6]
                             const int* __restrict__ conv_lut,         // [512,512]
                             const int* __restrict__ add_lut,          // [512,512]
                             const int* __restrict__ bias_lut,         // [512,6]
                             const int* __restrict__ relu_lut,         // [512]
                             unsigned short* __restrict__ x1)          // [B,14,14,6]
{
    __shared__ int sortbuf[25 * C1_BLOCK];
    int t   = threadIdx.x;
    int gid = blockIdx.x * C1_BLOCK + t;        // exact: 4096*196 = 64*12544
    int b   = gid / 196;
    int win = gid % 196;
    int oy  = win / 14, ox = win % 14;

    int patch[25];
    #pragma unroll
    for (int i = 0; i < 5; ++i)
        #pragma unroll
        for (int j = 0; j < 5; ++j)
            patch[i*5 + j] = sym[((size_t)b*32 + oy*2 + i)*32 + ox*2 + j];

    int* arr = sortbuf + t;
    for (int oc = 0; oc < 6; ++oc) {
        int cnt = 0;
        #pragma unroll
        for (int kk = 0; kk < 25; ++kk) {
            int w = c1w[kk*6 + oc];                       // uniform -> scalar load
            int v = conv_lut[patch[kk]*NK + w];
            int p = cnt;
            while (p > 0) {                               // insertion sort (ascending)
                int q = arr[(p-1)*C1_BLOCK];
                if (q <= v) break;
                arr[p*C1_BLOCK] = q;
                --p;
            }
            arr[p*C1_BLOCK] = v;
            ++cnt;
        }
        int acc = arr[0];
        #pragma unroll 1
        for (int i = 1; i < 25; ++i)
            acc = add_lut[arr[i*C1_BLOCK]*NK + acc];
        int s = bias_lut[acc*6 + oc];
        s = relu_lut[s];
        x1[(((size_t)b*14 + oy)*14 + ox)*6 + oc] = (unsigned short)s;
    }
}

// ---------------------------------------------------------------------------
// 3) Conv2 (k=5,s=2, 6ch -> 16ch): block = 4 windows x 16 channels.
//    Patch (150 syms/window) staged cooperatively in LDS; per-thread sort-150
//    column in LDS; writes real-valued feature (centroid) transposed to
//    [B,16,5,5] as f16 into feat_pad[B,416].
// ---------------------------------------------------------------------------
#define C2_BLOCK 64
#define KP1 416            // 400 padded to multiple of 32
__global__ void conv2_kernel(const unsigned short* __restrict__ x1,    // [B,14,14,6]
                             const int* __restrict__ c2w,              // [150,16]
                             const int* __restrict__ conv_lut,
                             const int* __restrict__ add_lut,
                             const int* __restrict__ bias_lut,         // [512,16]
                             const int* __restrict__ relu_lut,
                             const float* __restrict__ cent,           // [512]
                             _Float16* __restrict__ feat_pad)          // [B,416]
{
    __shared__ unsigned short patch[4][150];
    __shared__ int sortbuf[150 * C2_BLOCK];
    int t = threadIdx.x;
    int wgrp0 = blockIdx.x * 4;                    // windows: 4096*25 = 4*25600

    for (int idx = t; idx < 4*150; idx += C2_BLOCK) {
        int w = idx / 150, kk = idx % 150;
        int gwin = wgrp0 + w;
        int b = gwin / 25, win = gwin % 25;
        int oy = win / 5, ox = win % 5;
        int i = kk / 30, j = (kk % 30) / 6, c = kk % 6;
        patch[w][kk] = x1[(((size_t)b*14 + oy*2 + i)*14 + ox*2 + j)*6 + c];
    }
    __syncthreads();

    int w  = t >> 4, oc = t & 15;
    int gwin = wgrp0 + w;
    int b = gwin / 25, win = gwin % 25;

    int* arr = sortbuf + t;
    int cnt = 0;
    #pragma unroll 1
    for (int kk = 0; kk < 150; ++kk) {
        int wt = c2w[kk*16 + oc];
        int v  = conv_lut[(int)patch[w][kk]*NK + wt];
        int p = cnt;
        while (p > 0) {
            int q = arr[(p-1)*C2_BLOCK];
            if (q <= v) break;
            arr[p*C2_BLOCK] = q;
            --p;
        }
        arr[p*C2_BLOCK] = v;
        ++cnt;
    }
    int acc = arr[0];
    #pragma unroll 1
    for (int i = 1; i < 150; ++i)
        acc = add_lut[arr[i*C2_BLOCK]*NK + acc];
    int s = bias_lut[acc*16 + oc];
    s = relu_lut[s];
    feat_pad[(size_t)b*KP1 + oc*25 + win] = (_Float16)cent[s];
}

// ---------------------------------------------------------------------------
// 4) Weight packer: fcW [Nreal, Kreal] f32 -> Bpack [Kp/32][Np][32] f16,
//    Bpack[ks][n][kk] = W[n][ks*32+kk], zero padded.
// ---------------------------------------------------------------------------
__global__ void pack_w_kernel(const float* __restrict__ wsrc, _Float16* __restrict__ Bp,
                              int Kreal, int Nreal, int Np, int total)
{
    int idx = blockIdx.x * blockDim.x + threadIdx.x;
    if (idx >= total) return;
    int kk = idx & 31;
    int n  = (idx >> 5) % Np;
    int ks = (idx >> 5) / Np;
    int k  = (ks << 5) + kk;
    float v = (k < Kreal && n < Nreal) ? wsrc[n * Kreal + k] : 0.0f;
    Bp[idx] = (_Float16)v;
}

__global__ void zero_h16_kernel(_Float16* __restrict__ p, int n)
{
    int idx = blockIdx.x * blockDim.x + threadIdx.x;
    if (idx < n) p[idx] = (_Float16)0.0f;
}

// ---------------------------------------------------------------------------
// 5) WMMA GEMM: out[M,Np] = act(A[M,Kp] * B[Kp,Np] + bias).
//    One 16x16 tile per wave (4 waves/block). f16 inputs, f32 accumulate via
//    v_wmma_f32_16x16x32_f16. mode 0: sigmoid -> f16 (pad cols forced to 0);
//    mode 1: linear -> f32.
// ---------------------------------------------------------------------------
__global__ void gemm_wmma_kernel(const _Float16* __restrict__ A,
                                 const _Float16* __restrict__ Bp,
                                 const float* __restrict__ bias,
                                 _Float16* __restrict__ outH,
                                 float* __restrict__ outF,
                                 int M, int Kp, int Np, int Nreal, int mode)
{
    int wave = threadIdx.x >> 5;
    int lane = threadIdx.x & 31;
    int tile = blockIdx.x * 4 + wave;
    int tiles_n = Np >> 4;
    int tm = tile / tiles_n;
    int tn = tile % tiles_n;
    if (tm >= (M >> 4)) return;                  // uniform per wave

    int half = lane >> 4;                        // half-wave selects K sub-range
    int l16  = lane & 15;
    int m = (tm << 4) + l16;                     // A row for this lane
    int n = (tn << 4) + l16;                     // B/C column for this lane

    v8f acc = {};
    int KS = Kp >> 5;
    for (int ks = 0; ks < KS; ++ks) {
        int k0 = ks << 5;
        // A fragment: VGPR0-3 -> K = 8*half + [0,8); VGPR4-7 -> K = 16 + 8*half + [0,8)
        const _Float16* ap = A + (size_t)m * Kp + k0 + (half << 3);
        v8h alo = *(const v8h*)ap;
        v8h ahi = *(const v8h*)(ap + 16);
        v16h av;
        #pragma unroll
        for (int i = 0; i < 8; ++i) { av[i] = alo[i]; av[8 + i] = ahi[i]; }
        // B fragment: lane = column n, VGPR0-3 -> K = 8*half + [0,8); VGPR4-7 -> K = 16 + 8*half + [0,8)
        const _Float16* bp = Bp + ((size_t)(ks * Np + n) << 5) + (half << 3);
        v8h blo = *(const v8h*)bp;
        v8h bhi = *(const v8h*)(bp + 16);
        v16h bv;
        #pragma unroll
        for (int i = 0; i < 8; ++i) { bv[i] = blo[i]; bv[8 + i] = bhi[i]; }

        acc = __builtin_amdgcn_wmma_f32_16x16x32_f16(false, av, false, bv,
                                                     (short)0, acc, false, false);
    }

    float bn = (n < Nreal) ? bias[n] : 0.0f;
    #pragma unroll
    for (int r = 0; r < 8; ++r) {
        int mr = (tm << 4) + (half << 3) + r;    // C/D: VGPR r -> M = r + 8*half
        float v = acc[r] + bn;
        if (mode == 0) {
            float sg = 1.0f / (1.0f + __expf(-v));
            outH[(size_t)mr * Np + n] = (n < Nreal) ? (_Float16)sg : (_Float16)0.0f;
        } else {
            outF[(size_t)mr * Np + n] = v;
        }
    }
}

// ---------------------------------------------------------------------------
// 6) Softmax over 10 logits per row
// ---------------------------------------------------------------------------
__global__ void softmax10_kernel(const float* __restrict__ logits, float* __restrict__ out, int B)
{
    int r = blockIdx.x * blockDim.x + threadIdx.x;
    if (r >= B) return;
    float v[10];
    float mx = -1e30f;
    #pragma unroll
    for (int i = 0; i < 10; ++i) { v[i] = logits[r*16 + i]; mx = fmaxf(mx, v[i]); }
    float s = 0.0f;
    #pragma unroll
    for (int i = 0; i < 10; ++i) { v[i] = __expf(v[i] - mx); s += v[i]; }
    float inv = 1.0f / s;
    #pragma unroll
    for (int i = 0; i < 10; ++i) out[r*10 + i] = v[i] * inv;
}

// ---------------------------------------------------------------------------
extern "C" void kernel_launch(void* const* d_in, const int* in_sizes, int n_in,
                              void* d_out, int out_size, void* d_ws, size_t ws_size,
                              hipStream_t stream) {
    const float* x_bat    = (const float*)d_in[0];
    const float* cent     = (const float*)d_in[1];
    const int*   c1w      = (const int*)d_in[2];
    const int*   c2w      = (const int*)d_in[3];
    const int*   conv_lut = (const int*)d_in[4];
    const int*   add_lut  = (const int*)d_in[5];
    const int*   c1b      = (const int*)d_in[6];
    const int*   c2b      = (const int*)d_in[7];
    const int*   relu     = (const int*)d_in[8];
    const float* fc1w = (const float*)d_in[9];
    const float* fc1b = (const float*)d_in[10];
    const float* fc2w = (const float*)d_in[11];
    const float* fc2b = (const float*)d_in[12];
    const float* fc3w = (const float*)d_in[13];
    const float* fc3b = (const float*)d_in[14];
    float* out = (float*)d_out;

    char* ws = (char*)d_ws;
    size_t off = 0;
    auto carve = [&](size_t bytes) -> void* {
        void* p = ws + off;
        off = (off + bytes + 255) & ~(size_t)255;
        return p;
    };
    unsigned short* sym   = (unsigned short*)carve((size_t)BATCH*32*32*2);
    unsigned short* x1    = (unsigned short*)carve((size_t)BATCH*14*14*6*2);
    _Float16*       feat  = (_Float16*)carve((size_t)BATCH*KP1*2);       // [B,416]
    _Float16*       h1    = (_Float16*)carve((size_t)BATCH*128*2);       // [B,128]
    _Float16*       h2    = (_Float16*)carve((size_t)BATCH*96*2);        // [B,96]
    float*          lg    = (float*)carve((size_t)BATCH*16*4);           // [B,16]
    _Float16*       Bp1   = (_Float16*)carve((size_t)13*128*32*2);       // Kp=416
    _Float16*       Bp2   = (_Float16*)carve((size_t)4*96*32*2);         // Kp=128
    _Float16*       Bp3   = (_Float16*)carve((size_t)3*16*32*2);         // Kp=96

    // 1) discretize 4096*1024 pixels
    int npix = BATCH * 32 * 32;
    discretize_kernel<<<(npix + 255) / 256, 256, 0, stream>>>(x_bat, cent, sym, npix);

    // 2) conv1: 4096*196 windows / 64
    conv1_kernel<<<(BATCH * 196) / C1_BLOCK, C1_BLOCK, 0, stream>>>(
        sym, c1w, conv_lut, add_lut, c1b, relu, x1);

    // 3) zero feat pad, then conv2: 4096*25 windows, 4 windows/block
    int nfeat = BATCH * KP1;
    zero_h16_kernel<<<(nfeat + 255) / 256, 256, 0, stream>>>(feat, nfeat);
    conv2_kernel<<<(BATCH * 25) / 4, C2_BLOCK, 0, stream>>>(
        x1, c2w, conv_lut, add_lut, c2b, relu, cent, feat);

    // 4) pack FC weights
    int t1 = 13 * 128 * 32, t2 = 4 * 96 * 32, t3 = 3 * 16 * 32;
    pack_w_kernel<<<(t1 + 255) / 256, 256, 0, stream>>>(fc1w, Bp1, 400, 120, 128, t1);
    pack_w_kernel<<<(t2 + 255) / 256, 256, 0, stream>>>(fc2w, Bp2, 120,  84,  96, t2);
    pack_w_kernel<<<(t3 + 255) / 256, 256, 0, stream>>>(fc3w, Bp3,  84,  10,  16, t3);

    // 5) FC head via WMMA (16x16 tile per wave, 4 waves/block)
    // FC1: M=4096 Kp=416 Np=128 Nreal=120, sigmoid
    gemm_wmma_kernel<<<(256 * 8) / 4, 128, 0, stream>>>(feat, Bp1, fc1b, h1, nullptr,
                                                        BATCH, KP1, 128, 120, 0);
    // FC2: Kp=128 Np=96 Nreal=84, sigmoid
    gemm_wmma_kernel<<<(256 * 6) / 4, 128, 0, stream>>>(h1, Bp2, fc2b, h2, nullptr,
                                                        BATCH, 128, 96, 84, 0);
    // FC3: Kp=96 Np=16 Nreal=10, linear -> f32 logits
    gemm_wmma_kernel<<<(256 * 1) / 4, 128, 0, stream>>>(h2, Bp3, fc3b, nullptr, lg,
                                                        BATCH, 96, 16, 10, 1);

    // 6) softmax
    softmax10_kernel<<<(BATCH + 255) / 256, 256, 0, stream>>>(lg, out, BATCH);
}
